// TransformerLayer_31997506355694
// MI455X (gfx1250) — matrix operand order repaired and checked by reading, not compiled, not run on hardware
//
#include <hip/hip_runtime.h>
#include <hip/hip_bf16.h>
#include <math.h>
#include <stdint.h>

// ---------------------------------------------------------------------------
// MI455X (gfx1250) transformer layer + MoE.
// All large GEMMs via v_wmma_f32_16x16x32_f16 (f32->f16 tile conversion,
// f32 accumulation), 4 WMMAs per wave per K-step (32x32 per-wave tile).
// A-tiles DMA'd into LDS with the Tensor Data Mover (tensor_load_to_lds +
// s_wait_tensorcnt) when the builtin is available; cooperative-load fallback
// otherwise. wave32; fragment layouts per CDNA5 ISA 7.12.2.
// ---------------------------------------------------------------------------

typedef _Float16 half_t;
typedef __attribute__((ext_vector_type(16))) _Float16 v16h;
typedef __attribute__((ext_vector_type(8)))  _Float16 v8h;
typedef __attribute__((ext_vector_type(4)))  _Float16 v4h;
typedef __attribute__((ext_vector_type(8)))  float    v8f;
typedef __attribute__((ext_vector_type(4)))  float    v4f;
typedef __attribute__((ext_vector_type(4)))  unsigned int v4u;
typedef __attribute__((ext_vector_type(8)))  int      v8i;
typedef __attribute__((ext_vector_type(4)))  int      v4i;

#define D_MODEL 768
#define HD      64
#define NH      12
#define BB      4
#define SS      1024
#define NTOK    (BB*SS)      // 4096
#define FF      (D_MODEL*4)  // 3072
#define NEXP    16
#define CAP     320          // int(4096*1.25 // 16)

// Tensor Data Mover availability. This toolchain (clang-23 / therock HIP)
// declares the 6-arg form: (v4u g0, v8i g1, v4i g2, v4i g3, v8i, i32 cpol).
#if defined(__has_builtin)
#if __has_builtin(__builtin_amdgcn_tensor_load_to_lds)
#define USE_TDM 1
#endif
#if __has_builtin(__builtin_amdgcn_s_wait_tensorcnt)
#define HAVE_TENSORCNT_BUILTIN 1
#endif
#endif

__device__ __forceinline__ void wait_tensorcnt0() {
#if defined(HAVE_TENSORCNT_BUILTIN)
    __builtin_amdgcn_s_wait_tensorcnt(0);
#else
    asm volatile("s_wait_tensorcnt 0x0" ::: "memory");
#endif
}

__device__ __forceinline__ float gelu_exact(float v) {
    return 0.5f * v * (1.0f + erff(v * 0.70710678118654752f));
}

__device__ __forceinline__ v16h cat8(v8h lo, v8h hi) {
    return __builtin_shufflevector(lo, hi, 0,1,2,3,4,5,6,7,8,9,10,11,12,13,14,15);
}

__device__ __forceinline__ v8f wmma_f16(v16h a, v16h b, v8f c) {
    return __builtin_amdgcn_wmma_f32_16x16x32_f16(false, a, false, b, (short)0, c, false, false);
}

// ---------------------------------------------------------------------------
// LayerNorm (unbiased var, eps=1e-9). One 256-thread block per row of 768.
// ---------------------------------------------------------------------------
__global__ void __launch_bounds__(256)
ln_kernel(const float* __restrict__ x, const float* __restrict__ g,
          const float* __restrict__ b, half_t* __restrict__ outh,
          float* __restrict__ outf)
{
    const int row = blockIdx.x;
    const float* xr = x + (long)row * D_MODEL;
    __shared__ float red[256];
    const int tid = threadIdx.x;

    float s = 0.f;
    for (int i = tid; i < D_MODEL; i += 256) s += xr[i];
    red[tid] = s; __syncthreads();
    for (int off = 128; off > 0; off >>= 1) { if (tid < off) red[tid] += red[tid+off]; __syncthreads(); }
    const float mu = red[0] / (float)D_MODEL;
    __syncthreads();

    float s2 = 0.f;
    for (int i = tid; i < D_MODEL; i += 256) { float d = xr[i] - mu; s2 += d*d; }
    red[tid] = s2; __syncthreads();
    for (int off = 128; off > 0; off >>= 1) { if (tid < off) red[tid] += red[tid+off]; __syncthreads(); }
    const float var  = red[0] / (float)(D_MODEL - 1);
    const float rstd = rsqrtf(var + 1e-9f);

    for (int i = tid; i < D_MODEL; i += 256) {
        float y = (xr[i] - mu) * rstd * g[i] + b[i];
        outh[(long)row * D_MODEL + i] = (half_t)y;
        if (outf) outf[(long)row * D_MODEL + i] = y;
    }
}

// ---------------------------------------------------------------------------
// WMMA GEMM:  Out[M,N] = epi( A_f16[M,K] @ W_f32[K,N] )
// 256 threads = 8 waves in a 2x4 grid; block tile 64x128, K-step 32.
// Wave tile 32x32: 2 A-frags x 2 B-frags -> 4 v_wmma per K-step.
// A tile (64x32 f16, 40-half padded rows) loaded by the TDM; W tile converted
// f32->f16 and stored K-transposed so B-fragments are contiguous LDS reads.
// Batched over blockIdx.z (MoE experts). EPI: 1=bias, 2=gelu, 4=residual.
// ---------------------------------------------------------------------------
#define EPI_BIAS 1
#define EPI_GELU 2
#define EPI_RES  4

#define AP 40   // padded A row stride (halves): 32 data + 8 pad (TDM pad cfg)

template <typename OutT, int EPI>
__global__ void __launch_bounds__(256)
gemm_wmma(const half_t* __restrict__ A, const float* __restrict__ W,
          const float* __restrict__ bias, const float* __restrict__ resid,
          OutT* __restrict__ Out, int Mdim, int Ndim, int Kdim,
          long sA, long sW, long sO)
{
    const int z = blockIdx.z;
    A   += (long)z * sA;
    W   += (long)z * sW;
    Out += (long)z * sO;

    const int rowBase = blockIdx.x * 64;
    const int colBase = blockIdx.y * 128;

    __shared__ half_t As[64][AP];    // [row][k], 40-half stride
    __shared__ half_t Bs[128][40];   // [col][k], W transposed

    const int tid  = threadIdx.x;
    const int lane = tid & 31;
    const int wave = tid >> 5;       // 0..7
    const int wr   = wave >> 2;      // 0..1: rows wr*32
    const int wc   = wave & 3;       // 0..3: cols wc*32
    const int r    = lane & 15;
    const int hh   = lane >> 4;

    // W-tile loader mapping: lane = k (conflict-free transposed stores)
    const int wkk = tid & 31;          // k 0..31
    const int wn0 = (tid >> 5) * 16;   // n 0..112

#if defined(USE_TDM)
    const int wave_sc = __builtin_amdgcn_readfirstlane(wave);  // scalar wave id
    const unsigned int lds_a = (unsigned int)(uintptr_t)&As[0][0];
#else
    const int arow = tid >> 2;          // 0..63
    const int acol = (tid & 3) * 8;     // 0..24
#endif

    v8f acc00 = {}, acc01 = {}, acc10 = {}, acc11 = {};

    for (int k0 = 0; k0 < Kdim; k0 += 32) {
        __syncthreads();    // previous tile's fragment reads done

#if defined(USE_TDM)
        if (wave_sc == 0) {
            // Build 2-D D# (CDNA5 ISA ch.8): tile 32(k) x 64(rows), 2B elems,
            // LDS pad: +4 dwords after every 16 dwords (-> 40-half rows).
            const unsigned long long ga =
                (unsigned long long)(uintptr_t)(A + (long)rowBase * Kdim + k0);
            v4u g0;
            g0[0] = 1u;                                       // count=1
            g0[1] = lds_a;                                    // lds_addr
            g0[2] = (unsigned int)ga;                         // global_addr lo
            g0[3] = (unsigned int)((ga >> 32) & 0x1FFFFFFu)   // global_addr hi
                    | (2u << 30);                             // type=image
            v8i g1;
            g1[0] = (int)((1u << 16)        // data_size = 2 bytes
                    | (1u << 20)            // pad_enable
                    | (3u << 22)            // pad_interval = 16 dwords
                    | (3u << 25));          // pad_amount  = 4 dwords
            g1[1] = (int)(((unsigned)Kdim & 0xFFFFu) << 16);           // td0 lo
            g1[2] = (int)((((unsigned)Kdim >> 16) & 0xFFFFu)
                    | (((unsigned)Mdim & 0xFFFFu) << 16));             // td0 hi | td1 lo
            g1[3] = (int)((((unsigned)Mdim >> 16) & 0xFFFFu)
                    | (32u << 16));                                    // td1 hi | tile0=32
            g1[4] = 64;                                                // tile1=64
            g1[5] = Kdim;                                              // dim0 stride lo
            g1[6] = 0;
            g1[7] = 0;
            const v4i zz4 = {0, 0, 0, 0};
            const v8i zz8 = {0, 0, 0, 0, 0, 0, 0, 0};
            __builtin_amdgcn_tensor_load_to_lds(g0, g1, zz4, zz4, zz8, 0);
        }
#else
        {
            v8h av = *(const v8h*)(A + (long)(rowBase + arow) * Kdim + k0 + acol);
            *(v8h*)&As[arow][acol] = av;
        }
#endif
        // W tile: 32x128 f32 -> f16, transposed into Bs[col][k]
        {
            const float* wp = W + (long)(k0 + wkk) * Ndim + colBase + wn0;
            if (k0 + 32 < Kdim) __builtin_prefetch(wp + 32 * (long)Ndim, 0, 0);
            v4f w0 = *(const v4f*)(wp);
            v4f w1 = *(const v4f*)(wp + 4);
            v4f w2 = *(const v4f*)(wp + 8);
            v4f w3 = *(const v4f*)(wp + 12);
            #pragma unroll
            for (int i = 0; i < 4; ++i) Bs[wn0 + i][wkk]      = (half_t)w0[i];
            #pragma unroll
            for (int i = 0; i < 4; ++i) Bs[wn0 + 4 + i][wkk]  = (half_t)w1[i];
            #pragma unroll
            for (int i = 0; i < 4; ++i) Bs[wn0 + 8 + i][wkk]  = (half_t)w2[i];
            #pragma unroll
            for (int i = 0; i < 4; ++i) Bs[wn0 + 12 + i][wkk] = (half_t)w3[i];
        }
#if defined(USE_TDM)
        if (wave_sc == 0) wait_tensorcnt0();
#endif
        __syncthreads();

        // fragments: A lane r = row (K chunks {hh*8, 16+hh*8});
        //            B lane r = col (K chunks {hh*8, 16+hh*8})
        v8h l, h;
        l = *(const v8h*)&As[wr*32 + r][hh*8];
        h = *(const v8h*)&As[wr*32 + r][16 + hh*8];
        const v16h a0 = cat8(l, h);
        l = *(const v8h*)&As[wr*32 + 16 + r][hh*8];
        h = *(const v8h*)&As[wr*32 + 16 + r][16 + hh*8];
        const v16h a1 = cat8(l, h);
        l = *(const v8h*)&Bs[wc*32 + r][hh*8];
        h = *(const v8h*)&Bs[wc*32 + r][16 + hh*8];
        const v16h b0 = cat8(l, h);
        l = *(const v8h*)&Bs[wc*32 + 16 + r][hh*8];
        h = *(const v8h*)&Bs[wc*32 + 16 + r][16 + hh*8];
        const v16h b1 = cat8(l, h);

        acc00 = wmma_f16(a0, b0, acc00);
        acc01 = wmma_f16(a0, b1, acc01);
        acc10 = wmma_f16(a1, b0, acc10);
        acc11 = wmma_f16(a1, b1, acc11);
    }

    // Epilogue. D layout: lane r = col n, rows m = hh*8 + j.
    const int col0 = colBase + wc*32 + r;
    const int col1 = col0 + 16;
    float bv0 = 0.f, bv1 = 0.f;
    if (EPI & EPI_BIAS) { bv0 = bias[col0]; bv1 = bias[col1]; }
    #pragma unroll
    for (int j = 0; j < 8; ++j) {
        const int row0 = rowBase + wr*32 + hh*8 + j;
        const int row1 = row0 + 16;
        float v00 = acc00[j] + bv0, v01 = acc01[j] + bv1;
        float v10 = acc10[j] + bv0, v11 = acc11[j] + bv1;
        if (EPI & EPI_GELU) {
            v00 = gelu_exact(v00); v01 = gelu_exact(v01);
            v10 = gelu_exact(v10); v11 = gelu_exact(v11);
        }
        if (EPI & EPI_RES) {
            v00 += resid[(long)row0 * Ndim + col0];
            v01 += resid[(long)row0 * Ndim + col1];
            v10 += resid[(long)row1 * Ndim + col0];
            v11 += resid[(long)row1 * Ndim + col1];
        }
        Out[(long)row0 * Ndim + col0] = (OutT)v00;
        Out[(long)row0 * Ndim + col1] = (OutT)v01;
        Out[(long)row1 * Ndim + col0] = (OutT)v10;
        Out[(long)row1 * Ndim + col1] = (OutT)v11;
    }
}

// ---------------------------------------------------------------------------
// Causal attention, flash-style (8 WMMAs / key tile).
// ---------------------------------------------------------------------------
__global__ void __launch_bounds__(128)
attn_kernel(const half_t* __restrict__ Q, const half_t* __restrict__ K,
            const half_t* __restrict__ V, half_t* __restrict__ Oout)
{
    const int q0   = blockIdx.x * 64;
    const int head = blockIdx.y;
    const int bat  = blockIdx.z;
    const int tid  = threadIdx.x;
    const int wq   = tid >> 5;
    const int lane = tid & 31;
    const int r    = lane & 15;
    const int hh   = lane >> 4;
    const long base = ((long)bat * SS) * D_MODEL + head * HD;

    __shared__ half_t Kt[32][72];        // [key][d]
    __shared__ half_t Vt[64][40];        // [d][key]  (transposed)
    __shared__ half_t Pt[4][16][32];     // per-wave probabilities

    const half_t* qrowp = Q + base + (long)(q0 + wq*16 + r) * D_MODEL;
    v16h qf0, qf1;
    {
        v8h lo = *(const v8h*)(qrowp + hh*16);
        v8h hi = *(const v8h*)(qrowp + hh*16 + 8);
        qf0 = cat8(lo, hi);
        lo = *(const v8h*)(qrowp + 32 + hh*16);
        hi = *(const v8h*)(qrowp + 32 + hh*16 + 8);
        qf1 = cat8(lo, hi);
    }

    v8f o0 = {}, o1 = {}, o2 = {}, o3 = {};
    float mrow[8], lrow[8];
    #pragma unroll
    for (int j = 0; j < 8; ++j) { mrow[j] = -__builtin_inff(); lrow[j] = 0.f; }

    const int nkt  = q0/32 + 2;
    const int lkey = tid >> 2;           // 0..31
    const int ld0  = (tid & 3) * 16;     // 0,16,32,48

    for (int kt = 0; kt < nkt; ++kt) {
        __syncthreads();
        {
            const half_t* kp = K + base + (long)(kt*32 + lkey) * D_MODEL + ld0;
            v8h k0v = *(const v8h*)(kp);
            v8h k1v = *(const v8h*)(kp + 8);
            *(v8h*)&Kt[lkey][ld0]     = k0v;
            *(v8h*)&Kt[lkey][ld0 + 8] = k1v;
            const half_t* vp = V + base + (long)(kt*32 + lkey) * D_MODEL + ld0;
            v8h v0 = *(const v8h*)(vp);
            v8h v1 = *(const v8h*)(vp + 8);
            #pragma unroll
            for (int i = 0; i < 8; ++i) Vt[ld0 + i][lkey]     = v0[i];
            #pragma unroll
            for (int i = 0; i < 8; ++i) Vt[ld0 + 8 + i][lkey] = v1[i];
        }
        __syncthreads();

        v8f s0 = {}, s1 = {};
        {
            v8h l = *(const v8h*)&Kt[r][hh*16];
            v8h h = *(const v8h*)&Kt[r][hh*16 + 8];
            s0 = wmma_f16(qf0, cat8(l, h), s0);
            l = *(const v8h*)&Kt[r][32 + hh*16];
            h = *(const v8h*)&Kt[r][32 + hh*16 + 8];
            s0 = wmma_f16(qf1, cat8(l, h), s0);
            l = *(const v8h*)&Kt[16 + r][hh*16];
            h = *(const v8h*)&Kt[16 + r][hh*16 + 8];
            s1 = wmma_f16(qf0, cat8(l, h), s1);
            l = *(const v8h*)&Kt[16 + r][32 + hh*16];
            h = *(const v8h*)&Kt[16 + r][32 + hh*16 + 8];
            s1 = wmma_f16(qf1, cat8(l, h), s1);
        }

        const int qa   = q0 + wq*16 + hh*8;
        const int key0 = kt*32 + r;
        float corr[8];
        #pragma unroll
        for (int j = 0; j < 8; ++j) {
            const int qrow = qa + j;
            float a0 = (key0      <= qrow) ? s0[j] * 0.125f : -__builtin_inff();
            float a1 = (key0 + 16 <= qrow) ? s1[j] * 0.125f : -__builtin_inff();
            float t = fmaxf(a0, a1);
            #pragma unroll
            for (int off = 1; off < 16; off <<= 1) t = fmaxf(t, __shfl_xor(t, off));
            const float mn = fmaxf(mrow[j], t);
            const float c  = __expf(mrow[j] - mn);
            const float p0 = __expf(a0 - mn);
            const float p1 = __expf(a1 - mn);
            float ps = p0 + p1;
            #pragma unroll
            for (int off = 1; off < 16; off <<= 1) ps += __shfl_xor(ps, off);
            mrow[j] = mn;
            lrow[j] = lrow[j] * c + ps;
            corr[j] = c;
            Pt[wq][hh*8 + j][r]      = (half_t)p0;
            Pt[wq][hh*8 + j][16 + r] = (half_t)p1;
        }
        #pragma unroll
        for (int j = 0; j < 8; ++j) {
            o0[j] *= corr[j]; o1[j] *= corr[j]; o2[j] *= corr[j]; o3[j] *= corr[j];
        }
        __syncthreads();

        v8h plo = *(const v8h*)&Pt[wq][r][hh*16];
        v8h phi = *(const v8h*)&Pt[wq][r][hh*16 + 8];
        v16h pf = cat8(plo, phi);
        {
            v8h l, h;
            l = *(const v8h*)&Vt[ 0 + r][hh*16]; h = *(const v8h*)&Vt[ 0 + r][hh*16 + 8];
            o0 = wmma_f16(pf, cat8(l, h), o0);
            l = *(const v8h*)&Vt[16 + r][hh*16]; h = *(const v8h*)&Vt[16 + r][hh*16 + 8];
            o1 = wmma_f16(pf, cat8(l, h), o1);
            l = *(const v8h*)&Vt[32 + r][hh*16]; h = *(const v8h*)&Vt[32 + r][hh*16 + 8];
            o2 = wmma_f16(pf, cat8(l, h), o2);
            l = *(const v8h*)&Vt[48 + r][hh*16]; h = *(const v8h*)&Vt[48 + r][hh*16 + 8];
            o3 = wmma_f16(pf, cat8(l, h), o3);
        }
    }

    #pragma unroll
    for (int j = 0; j < 8; ++j) {
        const float inv = 1.f / lrow[j];
        half_t* orow = Oout + base + (long)(q0 + wq*16 + hh*8 + j) * D_MODEL;
        orow[ 0 + r] = (half_t)(o0[j] * inv);
        orow[16 + r] = (half_t)(o1[j] * inv);
        orow[32 + r] = (half_t)(o2[j] * inv);
        orow[48 + r] = (half_t)(o3[j] * inv);
    }
}

// ---------------------------------------------------------------------------
// Router second matmul (N=16: VALU dot), routing, scatter/gather, finalize.
// ---------------------------------------------------------------------------
__global__ void __launch_bounds__(256)
router2_kernel(const half_t* __restrict__ H1, const float* __restrict__ rw2,
               const float* __restrict__ rb2, float* __restrict__ rlog)
{
    const int t = blockIdx.x * 16 + (threadIdx.x >> 4);
    const int m = threadIdx.x & 15;
    const half_t* hp = H1 + (long)t * FF;
    float acc = rb2[m];
    for (int k = 0; k < FF; ++k) acc += (float)hp[k] * rw2[k * 16 + m];
    rlog[(long)t * 16 + m] = acc;
}

__global__ void __launch_bounds__(256)
route_kernel(const float* __restrict__ rlog, int* __restrict__ slot,
             int* __restrict__ eidx)
{
    const int tid = threadIdx.x;
    for (int t = tid; t < NTOK; t += 256) {
        const float* rp = rlog + (long)t * 16;
        float best = rp[0]; int bi = 0;
        for (int m = 1; m < 16; ++m) { float v = rp[m]; if (v > best) { best = v; bi = m; } }
        eidx[t] = bi;
    }
    __syncthreads();
    if (tid < NEXP) {
        int cnt = 0;
        for (int t = 0; t < NTOK; ++t) {
            if (eidx[t] == tid) {
                slot[t] = (cnt < CAP) ? (tid * CAP + cnt) : -1;
                ++cnt;
            }
        }
    }
}

__global__ void zero_half(half_t* p, long n)
{
    long i = (long)blockIdx.x * blockDim.x + threadIdx.x;
    const long stride = (long)gridDim.x * blockDim.x;
    for (; i < n; i += stride) p[i] = (half_t)0.f;
}

__global__ void __launch_bounds__(256)
scatter_kernel(const half_t* __restrict__ xn2h, const int* __restrict__ slot,
               half_t* __restrict__ Ein)
{
    const int t = blockIdx.x;
    const int s = slot[t];
    if (s < 0) return;
    const half_t* src = xn2h + (long)t * D_MODEL;
    half_t* dst = Ein + (long)s * D_MODEL;
    for (int i = threadIdx.x; i < D_MODEL; i += 256) dst[i] = src[i];
}

__global__ void __launch_bounds__(256)
final_kernel(const float* __restrict__ x1, const float* __restrict__ Eout,
             const float* __restrict__ xn2f, const int* __restrict__ slot,
             const float* __restrict__ rlog, float* __restrict__ out_x,
             float* __restrict__ out_rl)
{
    const int t = blockIdx.x;
    const int s = slot[t];
    const float* moe = (s >= 0) ? (Eout + (long)s * D_MODEL) : (xn2f + (long)t * D_MODEL);
    for (int i = threadIdx.x; i < D_MODEL; i += 256)
        out_x[(long)t * D_MODEL + i] = x1[(long)t * D_MODEL + i] + moe[i];
    if (threadIdx.x < 16) out_rl[t * 16 + threadIdx.x] = rlog[t * 16 + threadIdx.x];
}

// ---------------------------------------------------------------------------
extern "C" void kernel_launch(void* const* d_in, const int* in_sizes, int n_in,
                              void* d_out, int out_size, void* d_ws, size_t ws_size,
                              hipStream_t stream)
{
    const float* x     = (const float*)d_in[0];
    const float* ln1_g = (const float*)d_in[1];
    const float* ln1_b = (const float*)d_in[2];
    const float* Wq    = (const float*)d_in[3];
    const float* bq    = (const float*)d_in[4];
    const float* Wk    = (const float*)d_in[5];
    const float* bk    = (const float*)d_in[6];
    const float* Wv    = (const float*)d_in[7];
    const float* bv    = (const float*)d_in[8];
    const float* Wo    = (const float*)d_in[9];
    const float* bo    = (const float*)d_in[10];
    const float* ln2_g = (const float*)d_in[11];
    const float* ln2_b = (const float*)d_in[12];
    const float* rw1   = (const float*)d_in[13];
    const float* rb1   = (const float*)d_in[14];
    const float* rw2   = (const float*)d_in[15];
    const float* rb2   = (const float*)d_in[16];
    const float* mlp1  = (const float*)d_in[17];
    const float* mlp2  = (const float*)d_in[18];

    char* ws = (char*)d_ws;
    auto alloc = [&](size_t bytes) -> char* {
        char* p = ws;
        ws += (bytes + 255) & ~(size_t)255;
        return p;
    };

    half_t* xn1h  = (half_t*)alloc((size_t)NTOK * D_MODEL * 2);
    half_t* qh    = (half_t*)alloc((size_t)NTOK * D_MODEL * 2);
    half_t* kh    = (half_t*)alloc((size_t)NTOK * D_MODEL * 2);
    half_t* vh    = (half_t*)alloc((size_t)NTOK * D_MODEL * 2);
    half_t* attnh = (half_t*)alloc((size_t)NTOK * D_MODEL * 2);
    float*  x1    = (float*) alloc((size_t)NTOK * D_MODEL * 4);
    half_t* xn2h  = (half_t*)alloc((size_t)NTOK * D_MODEL * 2);
    float*  xn2f  = (float*) alloc((size_t)NTOK * D_MODEL * 4);
    half_t* h1    = (half_t*)alloc((size_t)NTOK * FF * 2);
    float*  rlog  = (float*) alloc((size_t)NTOK * 16 * 4);
    int*    eidx  = (int*)   alloc((size_t)NTOK * 4);
    int*    slot  = (int*)   alloc((size_t)NTOK * 4);
    half_t* Ein   = (half_t*)alloc((size_t)NEXP * CAP * D_MODEL * 2);
    half_t* Hmid  = (half_t*)alloc((size_t)NEXP * CAP * FF * 2);
    float*  Eout  = (float*) alloc((size_t)NEXP * CAP * D_MODEL * 4);

    // 1) LN1
    ln_kernel<<<NTOK, 256, 0, stream>>>(x, ln1_g, ln1_b, xn1h, nullptr);

    // 2) QKV projections (WMMA): block tile 64x128
    dim3 gdd(NTOK / 64, D_MODEL / 128, 1);
    gemm_wmma<half_t, EPI_BIAS><<<gdd, 256, 0, stream>>>(xn1h, Wq, bq, nullptr, qh, NTOK, D_MODEL, D_MODEL, 0, 0, 0);
    gemm_wmma<half_t, EPI_BIAS><<<gdd, 256, 0, stream>>>(xn1h, Wk, bk, nullptr, kh, NTOK, D_MODEL, D_MODEL, 0, 0, 0);
    gemm_wmma<half_t, EPI_BIAS><<<gdd, 256, 0, stream>>>(xn1h, Wv, bv, nullptr, vh, NTOK, D_MODEL, D_MODEL, 0, 0, 0);

    // 3) causal attention (WMMA flash)
    attn_kernel<<<dim3(SS / 64, NH, BB), 128, 0, stream>>>(qh, kh, vh, attnh);

    // 4) output projection + residual
    gemm_wmma<float, EPI_BIAS | EPI_RES><<<gdd, 256, 0, stream>>>(attnh, Wo, bo, x, x1, NTOK, D_MODEL, D_MODEL, 0, 0, 0);

    // 5) LN2
    ln_kernel<<<NTOK, 256, 0, stream>>>(x1, ln2_g, ln2_b, xn2h, xn2f);

    // 6) router FFN (gelu) + logits
    gemm_wmma<half_t, EPI_BIAS | EPI_GELU><<<dim3(NTOK / 64, FF / 128, 1), 256, 0, stream>>>(
        xn2h, rw1, rb1, nullptr, h1, NTOK, FF, D_MODEL, 0, 0, 0);
    router2_kernel<<<NTOK / 16, 256, 0, stream>>>(h1, rw2, rb2, rlog);

    // 7) routing + capacity
    route_kernel<<<1, 256, 0, stream>>>(rlog, slot, eidx);

    // 8) scatter tokens into expert buffers
    zero_half<<<512, 256, 0, stream>>>(Ein, (long)NEXP * CAP * D_MODEL);
    scatter_kernel<<<NTOK, 256, 0, stream>>>(xn2h, slot, Ein);

    // 9) expert GEMMs (batched over blockIdx.z = expert)
    gemm_wmma<half_t, EPI_GELU><<<dim3(CAP / 64, FF / 128, NEXP), 256, 0, stream>>>(
        Ein, mlp1, nullptr, nullptr, Hmid, CAP, FF, D_MODEL,
        (long)CAP * D_MODEL, (long)D_MODEL * FF, (long)CAP * FF);
    gemm_wmma<float, 0><<<dim3(CAP / 64, D_MODEL / 128, NEXP), 256, 0, stream>>>(
        Hmid, mlp2, nullptr, nullptr, Eout, CAP, D_MODEL, FF,
        (long)CAP * FF, (long)FF * D_MODEL, (long)CAP * D_MODEL);

    // 10) gather + residual + router logits out
    float* out_x  = (float*)d_out;
    float* out_rl = (float*)d_out + (long)NTOK * D_MODEL;
    final_kernel<<<NTOK, 256, 0, stream>>>(x1, Eout, xn2f, slot, rlog, out_x, out_rl);
}
